// SparseMNISTCNN_79731772883542
// MI455X (gfx1250) — matrix operand order repaired and checked by reading, not compiled, not run on hardware
//
#include <hip/hip_runtime.h>
#include <hip/hip_bf16.h>
#include <math.h>

typedef __attribute__((ext_vector_type(16))) __bf16 v16bf;
typedef __attribute__((ext_vector_type(8)))  float  v8f;

#define BATCH 4096
#define C1_OUT 64
#define CLEN   9216   // 64 * 12 * 12
#define C1K    400
#define NFC    1024
#define KFC    100
#define TEN    10

__device__ __forceinline__ float  bf2f(__bf16 v) { return (float)v; }
__device__ __forceinline__ __bf16 f2bf(float f)  { return (__bf16)f; }

union FragBf {            // lets us load fragments as 2x b128 from LDS
  v16bf v;
  uint4 q[2];
};

// ---------------------------------------------------------------------------
// K0: conv5x5 (valid) + bias + maxpool2x2, output bf16 [B, 9216]
// ---------------------------------------------------------------------------
__global__ void k0_conv_pool(const float* __restrict__ x,
                             const float* __restrict__ cw,
                             const float* __restrict__ cb,
                             __bf16* __restrict__ Y) {
  int gid = blockIdx.x * 256 + threadIdx.x;
  if (gid >= BATCH * CLEN) return;
  int b   = gid / CLEN;
  int rem = gid % CLEN;
  int c   = rem / 144;
  int s   = rem % 144;
  int ph = s / 12, pw = s % 12;
  const float* xb = x + (size_t)b * 784;
  const float* wc = cw + c * 25;
  float patch[6][6];
  int h0 = ph * 2, w0 = pw * 2;
#pragma unroll
  for (int i = 0; i < 6; ++i)
#pragma unroll
    for (int j = 0; j < 6; ++j)
      patch[i][j] = xb[(h0 + i) * 28 + (w0 + j)];
  float bias = cb[c];
  float m = -INFINITY;
#pragma unroll
  for (int dh = 0; dh < 2; ++dh)
#pragma unroll
    for (int dw = 0; dw < 2; ++dw) {
      float sum = bias;
#pragma unroll
      for (int i = 0; i < 5; ++i)
#pragma unroll
        for (int j = 0; j < 5; ++j)
          sum = fmaf(patch[dh + i][dw + j], wc[i * 5 + j], sum);
      m = fmaxf(m, sum);
    }
  Y[gid] = f2bf(m);
}

// ---------------------------------------------------------------------------
// K1: per-row k-th largest boosted value via bisection (top-400 of 9216)
// ---------------------------------------------------------------------------
__global__ void k1_cnn_thresh(const __bf16* __restrict__ Y,
                              const float* __restrict__ duty_cnn,
                              float* __restrict__ thr) {
  __shared__ float boostS[C1_OUT];
  __shared__ float red[256];
  __shared__ int cnt;
  int b = blockIdx.x;
  int t = threadIdx.x;
  if (t < C1_OUT) boostS[t] = expf((400.0f / 9216.0f) - duty_cnn[t]);
  __syncthreads();
  const __bf16* row = Y + (size_t)b * CLEN;
  float lmax = -INFINITY, lmin = INFINITY;
  for (int i = t; i < CLEN; i += 256) {
    float v = bf2f(row[i]) * boostS[i / 144];
    lmax = fmaxf(lmax, v); lmin = fminf(lmin, v);
  }
  red[t] = lmax; __syncthreads();
  for (int s = 128; s > 0; s >>= 1) { if (t < s) red[t] = fmaxf(red[t], red[t + s]); __syncthreads(); }
  float hi = red[0]; __syncthreads();
  red[t] = lmin; __syncthreads();
  for (int s = 128; s > 0; s >>= 1) { if (t < s) red[t] = fminf(red[t], red[t + s]); __syncthreads(); }
  float lo = red[0]; __syncthreads();
  for (int it = 0; it < 26; ++it) {
    float mid = 0.5f * (lo + hi);
    if (t == 0) cnt = 0;
    __syncthreads();
    int c = 0;
    for (int i = t; i < CLEN; i += 256)
      c += (bf2f(row[i]) * boostS[i / 144] >= mid) ? 1 : 0;
    atomicAdd(&cnt, c);
    __syncthreads();
    int total = cnt;
    __syncthreads();
    if (total >= C1K) lo = mid; else hi = mid;
  }
  if (t == 0) thr[b] = lo;
}

// ---------------------------------------------------------------------------
// K2: apply k-winner mask in place (keeps raw value where boosted >= thr)
// ---------------------------------------------------------------------------
__global__ void k2_cnn_apply(__bf16* __restrict__ Y,
                             const float* __restrict__ duty_cnn,
                             const float* __restrict__ thr) {
  int gid = blockIdx.x * 256 + threadIdx.x;
  if (gid >= BATCH * CLEN) return;
  int b = gid / CLEN;
  int i = gid % CLEN;
  int c = i / 144;
  float boost = expf((400.0f / 9216.0f) - duty_cnn[c]);
  if (bf2f(Y[gid]) * boost < thr[b]) Y[gid] = f2bf(0.0f);
}

// ---------------------------------------------------------------------------
// K3: masked fc1 weights -> bf16 B matrix in WMMA-fragment-major layout:
//     Bw[(k/32)*(NFC*32) + n*32 + (k%32)]
// Input fc1_w is [N, K] row-major (k contiguous) so no transpose is needed:
// each thread repacks 8 consecutive k of one n. Fully coalesced reads,
// 16B-contiguous writes.
// ---------------------------------------------------------------------------
__global__ void k3_prep_weights(const float* __restrict__ fc1_w,
                                const float* __restrict__ mraw,
                                __bf16* __restrict__ Bw) {
  int q = blockIdx.x * 256 + threadIdx.x;      // one 8-element k-chunk
  if (q >= NFC * (CLEN / 8)) return;
  int j = q / (CLEN / 8);                      // n index
  int i = (q % (CLEN / 8)) * 8;                // k index (multiple of 8)
  size_t g = (size_t)j * CLEN + i;
  union { __bf16 h[8]; uint4 q4; } tmp;
#pragma unroll
  for (int e = 0; e < 8; ++e) {
    float w = fc1_w[g + e];
    float m = mraw[g + e];
    tmp.h[e] = f2bf(m < 0.5f ? w : 0.0f);
  }
  size_t o = (size_t)(i >> 5) * (NFC * 32) + (size_t)j * 32 + (i & 31);
  *(uint4*)(Bw + o) = tmp.q4;
}

// ---------------------------------------------------------------------------
// K4: fc1 GEMM  H[4096,1024] = A[4096,9216](bf16) x B(bf16, frag-major) + bias
// 128x64 block tile, 8 waves -> 32x32 wave tiles, K-step 32,
// double-buffered LDS (one barrier per K-step), all LDS traffic in b128.
// ---------------------------------------------------------------------------
__global__ __launch_bounds__(256)
void k4_fc1_wmma(const __bf16* __restrict__ A,
                 const __bf16* __restrict__ Bw,
                 const float* __restrict__ fc1_b,
                 float* __restrict__ H) {
  __shared__ __align__(16) __bf16 As[2][128][32];
  __shared__ __align__(16) __bf16 Bs[2][64 * 32];
  int t = threadIdx.x;
  int lane = t & 31, wid = t >> 5;
  int l15 = lane & 15, lhi = lane >> 4;
  int wm = wid & 3, wn = wid >> 2;             // 4x2 wave grid
  int m_base = blockIdx.y * 128;
  int n_base = blockIdx.x * 64;

  v8f acc[2][2];
#pragma unroll
  for (int mi = 0; mi < 2; ++mi)
#pragma unroll
    for (int ni = 0; ni < 2; ++ni) { v8f z = {}; acc[mi][ni] = z; }

  // staging map: A tile 128x32 bf16 = 512 uint4 (2/thread); B panel = 256 uint4
  int ar0 = (t * 2)     >> 2, ac0 = ((t * 2)     & 3) * 8;
  int ar1 = (t * 2 + 1) >> 2, ac1 = ((t * 2 + 1) & 3) * 8;
  const __bf16* Ag0 = A + (size_t)(m_base + ar0) * CLEN + ac0;
  const __bf16* Ag1 = A + (size_t)(m_base + ar1) * CLEN + ac1;
  const __bf16* Bg  = Bw + (size_t)n_base * 32 + t * 8;   // + panel*(NFC*32)

  uint4 ra0 = *(const uint4*)Ag0;
  uint4 ra1 = *(const uint4*)Ag1;
  uint4 rb  = *(const uint4*)Bg;
  *(uint4*)&As[0][ar0][ac0] = ra0;
  *(uint4*)&As[0][ar1][ac1] = ra1;
  *(uint4*)&Bs[0][t * 8]    = rb;
  __syncthreads();

  int buf = 0;
  for (int k0 = 0; k0 < CLEN; k0 += 32) {
    bool has_next = (k0 + 32) < CLEN;
    if (has_next) {                                       // loads for next tile
      ra0 = *(const uint4*)(Ag0 + k0 + 32);
      ra1 = *(const uint4*)(Ag1 + k0 + 32);
      rb  = *(const uint4*)(Bg + (size_t)((k0 >> 5) + 1) * (NFC * 32));
      if (k0 + 64 < CLEN) {                               // global_prefetch_b8
        __builtin_prefetch((const void*)(Ag0 + k0 + 64), 0, 3);
        __builtin_prefetch((const void*)(Bg + (size_t)((k0 >> 5) + 2) * (NFC * 32)), 0, 3);
      }
    }

    // Fragments: each = two contiguous 16B LDS chunks (ds_load_b128 pairs)
    FragBf af[2], bfu[2];
#pragma unroll
    for (int mi = 0; mi < 2; ++mi) {
      int mrow = wm * 32 + mi * 16 + l15;
      af[mi].q[0] = *(const uint4*)&As[buf][mrow][lhi * 8];
      af[mi].q[1] = *(const uint4*)&As[buf][mrow][16 + lhi * 8];
    }
#pragma unroll
    for (int ni = 0; ni < 2; ++ni) {
      int ncol = wn * 32 + ni * 16 + l15;
      bfu[ni].q[0] = *(const uint4*)&Bs[buf][ncol * 32 + lhi * 8];
      bfu[ni].q[1] = *(const uint4*)&Bs[buf][ncol * 32 + 16 + lhi * 8];
    }
#pragma unroll
    for (int mi = 0; mi < 2; ++mi)
#pragma unroll
      for (int ni = 0; ni < 2; ++ni)
        acc[mi][ni] = __builtin_amdgcn_wmma_f32_16x16x32_bf16(
            false, af[mi].v, false, bfu[ni].v, (short)0, acc[mi][ni], false, false);

    if (has_next) {                                       // fill other buffer
      *(uint4*)&As[buf ^ 1][ar0][ac0] = ra0;
      *(uint4*)&As[buf ^ 1][ar1][ac1] = ra1;
      *(uint4*)&Bs[buf ^ 1][t * 8]    = rb;
      __syncthreads();
      buf ^= 1;
    }
  }

  // epilogue: C/D layout (VGPR r: M = r + 8*hi, N = lane&15) + bias
#pragma unroll
  for (int mi = 0; mi < 2; ++mi)
#pragma unroll
    for (int ni = 0; ni < 2; ++ni) {
      int col = n_base + wn * 32 + ni * 16 + l15;
      float bias = fc1_b[col];
#pragma unroll
      for (int r = 0; r < 8; ++r) {
        int row = m_base + wm * 32 + mi * 16 + r + lhi * 8;
        H[(size_t)row * NFC + col] = acc[mi][ni][r] + bias;
      }
    }
}

// ---------------------------------------------------------------------------
// K5: fused fc k-winners (top-100 of 1024, bisection) + fc2 + log_softmax
// ---------------------------------------------------------------------------
__global__ void k5_head(const float* __restrict__ H,
                        const float* __restrict__ duty_fc,
                        const float* __restrict__ fc2_w,
                        const float* __restrict__ fc2_b,
                        float* __restrict__ out) {
  __shared__ float hrow[NFC];
  __shared__ float bb[NFC];
  __shared__ float red[256];
  __shared__ int cnt;
  __shared__ float logits[TEN];
  int b = blockIdx.x, t = threadIdx.x;
#pragma unroll
  for (int r = 0; r < 4; ++r) {
    int j = t + r * 256;
    float v = H[(size_t)b * NFC + j];
    hrow[j] = v;
    bb[j] = v * expf((100.0f / 1024.0f) - duty_fc[j]);
  }
  __syncthreads();
  float lmax = -INFINITY, lmin = INFINITY;
#pragma unroll
  for (int r = 0; r < 4; ++r) {
    float v = bb[t + r * 256];
    lmax = fmaxf(lmax, v); lmin = fminf(lmin, v);
  }
  red[t] = lmax; __syncthreads();
  for (int s = 128; s > 0; s >>= 1) { if (t < s) red[t] = fmaxf(red[t], red[t + s]); __syncthreads(); }
  float hi = red[0]; __syncthreads();
  red[t] = lmin; __syncthreads();
  for (int s = 128; s > 0; s >>= 1) { if (t < s) red[t] = fminf(red[t], red[t + s]); __syncthreads(); }
  float lo = red[0]; __syncthreads();
  for (int it = 0; it < 26; ++it) {
    float mid = 0.5f * (lo + hi);
    if (t == 0) cnt = 0;
    __syncthreads();
    int c = 0;
#pragma unroll
    for (int r = 0; r < 4; ++r) c += (bb[t + r * 256] >= mid) ? 1 : 0;
    atomicAdd(&cnt, c);
    __syncthreads();
    int total = cnt;
    __syncthreads();
    if (total >= KFC) lo = mid; else hi = mid;
  }
  float thr = lo;
#pragma unroll
  for (int r = 0; r < 4; ++r) {
    int j = t + r * 256;
    if (bb[j] < thr) hrow[j] = 0.0f;
  }
  __syncthreads();
  float p[TEN];
#pragma unroll
  for (int c = 0; c < TEN; ++c) p[c] = 0.0f;
#pragma unroll
  for (int r = 0; r < 4; ++r) {
    int j = t + r * 256;
    float v = hrow[j];
#pragma unroll
    for (int c = 0; c < TEN; ++c) p[c] = fmaf(v, fc2_w[c * NFC + j], p[c]);
  }
  for (int c = 0; c < TEN; ++c) {       // deterministic tree reductions
    red[t] = p[c]; __syncthreads();
    for (int s = 128; s > 0; s >>= 1) { if (t < s) red[t] += red[t + s]; __syncthreads(); }
    if (t == 0) logits[c] = red[0] + fc2_b[c];
    __syncthreads();
  }
  if (t == 0) {
    float m = -INFINITY;
    for (int c = 0; c < TEN; ++c) m = fmaxf(m, logits[c]);
    float s = 0.0f;
    for (int c = 0; c < TEN; ++c) s += expf(logits[c] - m);
    float lse = m + logf(s);
    for (int c = 0; c < TEN; ++c) out[(size_t)b * TEN + c] = logits[c] - lse;
  }
}

// ---------------------------------------------------------------------------
static inline size_t align256(size_t x) { return (x + 255) & ~(size_t)255; }

extern "C" void kernel_launch(void* const* d_in, const int* in_sizes, int n_in,
                              void* d_out, int out_size, void* d_ws, size_t ws_size,
                              hipStream_t stream) {
  const float* x        = (const float*)d_in[0];
  const float* c1_w     = (const float*)d_in[1];
  const float* c1_b     = (const float*)d_in[2];
  const float* duty_cnn = (const float*)d_in[3];
  const float* fc1_w    = (const float*)d_in[4];
  const float* fc1_mask = (const float*)d_in[5];
  const float* fc1_b    = (const float*)d_in[6];
  const float* duty_fc  = (const float*)d_in[7];
  const float* fc2_w    = (const float*)d_in[8];
  const float* fc2_b    = (const float*)d_in[9];
  float* out = (float*)d_out;

  char* ws = (char*)d_ws;
  size_t off = 0;
  __bf16* Y  = (__bf16*)(ws + off); off += align256((size_t)BATCH * CLEN * sizeof(__bf16));
  __bf16* Bw = (__bf16*)(ws + off); off += align256((size_t)CLEN * NFC * sizeof(__bf16));
  float*  H  = (float*) (ws + off); off += align256((size_t)BATCH * NFC * sizeof(float));
  float*  thr= (float*) (ws + off); off += align256((size_t)BATCH * sizeof(float));
  (void)off; (void)ws_size; (void)in_sizes; (void)n_in; (void)out_size;

  int nconv = BATCH * CLEN;
  k0_conv_pool<<<(nconv + 255) / 256, 256, 0, stream>>>(x, c1_w, c1_b, Y);
  k1_cnn_thresh<<<BATCH, 256, 0, stream>>>(Y, duty_cnn, thr);
  k2_cnn_apply<<<(nconv + 255) / 256, 256, 0, stream>>>(Y, duty_cnn, thr);
  k3_prep_weights<<<(NFC * (CLEN / 8) + 255) / 256, 256, 0, stream>>>(fc1_w, fc1_mask, Bw);
  k4_fc1_wmma<<<dim3(NFC / 64, BATCH / 128), 256, 0, stream>>>(Y, Bw, fc1_b, H);
  k5_head<<<BATCH, 256, 0, stream>>>(H, duty_fc, fc2_w, fc2_b, out);
}
